// SubGraphProcess_55070070669488
// MI455X (gfx1250) — matrix operand order repaired and checked by reading, not compiled, not run on hardware
//
#include <hip/hip_runtime.h>
#include <hip/hip_bf16.h>

#define N_NODES 20000
#define N_EDGES 100000
#define DIM 512
#define EFD 128
#define NH 16
#define HEADD 32
#define HLD 520            // f32 row stride of hidden buffer (16*520*4 = 33280 B)
#define SMEM_BYTES 33280   // max(16*1040*2, 16*520*4)

typedef unsigned short u16t;
typedef __attribute__((ext_vector_type(16))) __bf16 v16bf;
typedef __attribute__((ext_vector_type(8)))  float  v8f;

// ---------- small helpers ----------
__device__ __forceinline__ u16t f2bf(float x) {
  unsigned u = __float_as_uint(x);
  unsigned r = (u + 0x7fffu + ((u >> 16) & 1u)) >> 16;   // round-to-nearest-even
  return (u16t)r;
}
__device__ __forceinline__ float bf2f(u16t b) {
  return __uint_as_float(((unsigned)b) << 16);
}
__device__ __forceinline__ ushort4 f2bf4(float4 x) {
  return make_ushort4(f2bf(x.x), f2bf(x.y), f2bf(x.z), f2bf(x.w));
}
__device__ __forceinline__ void atomicMaxF(float* a, float v) {
  if (v >= 0.f) atomicMax((int*)a, __float_as_int(v));
  else          atomicMin((unsigned int*)a, __float_as_uint(v));
}

// ---------- WMMA fragment loaders ----------
// A (16x32 bf16, MxK): lanes 0-15 rows M=lane, K {0..7,16..23}; lanes 16-31 same rows, K +8.
// Element pairs (2p,2p+1) are consecutive K -> merged LDS loads.
__device__ __forceinline__ v16bf load_a_frag(const u16t* A, int ldk, int lane, int k0) {
  union { v16bf v; u16t s[16]; } u;
  int m   = lane & 15;
  int hi8 = (lane & 16) ? 8 : 0;
  const u16t* row = A + m * ldk + k0 + hi8;
#pragma unroll
  for (int p = 0; p < 8; ++p) {
    int k = 2 * p + ((p >= 4) ? 8 : 0);
    u.s[2 * p]     = row[k];
    u.s[2 * p + 1] = row[k + 1];
  }
  return u.v;
}

// B (32x16 bf16, KxN): pre-packed so each lane's 16 values are contiguous (32 B -> 2x b128).
__device__ __forceinline__ v16bf load_b_frag(const u16t* Wp, int ntn, int kt, int nt, int lane) {
  const u16t* p = Wp + ((size_t)((kt * ntn + nt) * 32 + lane)) * 16;
  union { v16bf v; u16t s[16]; } u;
#pragma unroll
  for (int i = 0; i < 16; ++i) u.s[i] = p[i];
  return u.v;
}

// ---------- layer-1 WMMA accumulation: 16 rows x 512 cols, 8 waves, 4 n-tiles/wave ----------
template <int K1, int LDX>
__device__ __forceinline__ void mlp_layer1(const u16t* Xs, const u16t* W1p,
                                           int lane, int wave, v8f (&acc)[4]) {
#pragma unroll
  for (int t = 0; t < 4; ++t)
#pragma unroll
    for (int v = 0; v < 8; ++v) acc[t][v] = 0.f;
  for (int kt = 0; kt < K1 / 32; ++kt) {
    v16bf a = load_a_frag(Xs, LDX, lane, kt * 32);
#pragma unroll
    for (int t = 0; t < 4; ++t) {
      v16bf b = load_b_frag(W1p, 32, kt, wave * 4 + t, lane);
      acc[t] = __builtin_amdgcn_wmma_f32_16x16x32_bf16(false, a, false, b,
                                                       (short)0, acc[t], false, false);
    }
  }
}

// ---------- +b1 -> LN*g+be -> relu -> bf16 -> @W2 + b2 ; result in hbuf ----------
// hbuf may alias the Xs staging buffer (first barrier separates the phases).
__device__ __forceinline__ void mlp_finish(
    v8f (&acc)[4], float* hbuf, u16t* abuf,
    const float* b1, const float* g, const float* be,
    const u16t* W2p, const float* b2, int lane, int wave)
{
  __syncthreads();  // all Xs reads (possibly from multiple layer1 passes) done
  int mb = (lane & 16) ? 8 : 0;
  int nc = lane & 15;
#pragma unroll
  for (int t = 0; t < 4; ++t) {
    int n = wave * 64 + t * 16 + nc;
    float bb = b1[n];
#pragma unroll
    for (int v = 0; v < 8; ++v) hbuf[(mb + v) * HLD + n] = acc[t][v] + bb;
  }
  __syncthreads();

  // LayerNorm + affine + relu -> bf16 in abuf. Wave w handles rows 2w, 2w+1.
#pragma unroll
  for (int rr = 0; rr < 2; ++rr) {
    int m = wave * 2 + rr;
    float xv[16];
    float s = 0.f, s2 = 0.f;
#pragma unroll
    for (int jj = 0; jj < 4; ++jj) {
      float4 x4 = *(const float4*)&hbuf[m * HLD + jj * 128 + lane * 4];
      xv[jj * 4 + 0] = x4.x; xv[jj * 4 + 1] = x4.y;
      xv[jj * 4 + 2] = x4.z; xv[jj * 4 + 3] = x4.w;
      s  += x4.x + x4.y + x4.z + x4.w;
      s2 += x4.x * x4.x + x4.y * x4.y + x4.z * x4.z + x4.w * x4.w;
    }
#pragma unroll
    for (int o = 16; o >= 1; o >>= 1) { s += __shfl_xor(s, o, 32); s2 += __shfl_xor(s2, o, 32); }
    float mu   = s * (1.f / 512.f);
    float rstd = rsqrtf(s2 * (1.f / 512.f) - mu * mu + 1e-5f);
#pragma unroll
    for (int jj = 0; jj < 4; ++jj) {
      int j = jj * 128 + lane * 4;
      float4 g4  = *(const float4*)&g[j];
      float4 be4 = *(const float4*)&be[j];
      ushort4 o4;
      o4.x = f2bf(fmaxf((xv[jj * 4 + 0] - mu) * rstd * g4.x + be4.x, 0.f));
      o4.y = f2bf(fmaxf((xv[jj * 4 + 1] - mu) * rstd * g4.y + be4.y, 0.f));
      o4.z = f2bf(fmaxf((xv[jj * 4 + 2] - mu) * rstd * g4.z + be4.z, 0.f));
      o4.w = f2bf(fmaxf((xv[jj * 4 + 3] - mu) * rstd * g4.w + be4.w, 0.f));
      *(ushort4*)&abuf[m * 512 + j] = o4;
    }
  }
  __syncthreads();

  // layer 2: K = 512 (reuse acc)
#pragma unroll
  for (int t = 0; t < 4; ++t)
#pragma unroll
    for (int v = 0; v < 8; ++v) acc[t][v] = 0.f;
  for (int kt = 0; kt < 16; ++kt) {
    v16bf a = load_a_frag(abuf, 512, lane, kt * 32);
#pragma unroll
    for (int t = 0; t < 4; ++t) {
      v16bf b = load_b_frag(W2p, 32, kt, wave * 4 + t, lane);
      acc[t] = __builtin_amdgcn_wmma_f32_16x16x32_bf16(false, a, false, b,
                                                       (short)0, acc[t], false, false);
    }
  }
  __syncthreads();
#pragma unroll
  for (int t = 0; t < 4; ++t) {
    int n = wave * 64 + t * 16 + nc;
    float bb = b2[n];
#pragma unroll
    for (int v = 0; v < 8; ++v) hbuf[(mb + v) * HLD + n] = acc[t][v] + bb;
  }
  __syncthreads();
}

// ---------- weight pre-pack: f32 KxN row-major -> per-lane B-fragment layout, bf16 ----------
__global__ void pack_w_kernel(const float* __restrict__ W, u16t* __restrict__ Wp, int K, int N) {
  long long i = (long long)blockIdx.x * blockDim.x + threadIdx.x;
  long long total = (long long)K * N;
  if (i >= total) return;
  int e    = (int)(i & 15);
  int lane = (int)((i >> 4) & 31);
  long long tile = i >> 9;
  int Nt = N >> 4;
  int nt = (int)(tile % Nt);
  int kt = (int)(tile / Nt);
  int k = kt * 32 + ((lane & 16) >> 1) + (e & 7) + ((e & 8) << 1);
  int n = nt * 16 + (lane & 15);
  Wp[i] = f2bf(W[(size_t)k * N + n]);
}

// ---------- MLP kernels ----------
__global__ __launch_bounds__(256) void q_mlp_kernel(
    const float* __restrict__ h,
    const u16t* W1p, const float* b1, const float* g, const float* be,
    const u16t* W2p, const float* b2,
    float* __restrict__ qout)
{
  __shared__ __align__(16) unsigned char smem[SMEM_BYTES];
  __shared__ u16t abuf[16 * 512];
  u16t*  Xs   = (u16t*)smem;
  float* hbuf = (float*)smem;
  int tid  = threadIdx.x;
  int lane = tid & 31, wave = tid >> 5;
  int r0   = blockIdx.x * 16;

  for (int i = tid; i < 16 * 128; i += 256) {   // 512/4 vec4 per row
    int m = i >> 7, j = (i & 127) * 4;
    float4 x4 = *(const float4*)&h[(size_t)(r0 + m) * 512 + j];
    *(ushort4*)&Xs[m * 528 + j] = f2bf4(x4);
  }
  __syncthreads();
  v8f acc[4];
  mlp_layer1<512, 528>(Xs, W1p, lane, wave, acc);
  mlp_finish(acc, hbuf, abuf, b1, g, be, W2p, b2, lane, wave);
  for (int i = tid; i < 16 * 128; i += 256) {
    int m = i >> 7, j = (i & 127) * 4;
    *(float4*)&qout[(size_t)(r0 + m) * 512 + j] = *(const float4*)&hbuf[m * HLD + j];
  }
}

__global__ __launch_bounds__(256) void kv_mlp_kernel(
    const float* __restrict__ h, const float* __restrict__ ef, const int* __restrict__ src,
    const u16t* kW1p, const float* kb1, const float* kg, const float* kbe,
    const u16t* kW2p, const float* kb2,
    const u16t* vW1p, const float* vb1, const float* vg, const float* vbe,
    const u16t* vW2p, const float* vb2,
    u16t* __restrict__ kout, u16t* __restrict__ vout)
{
  __shared__ __align__(16) unsigned char smem[SMEM_BYTES];
  __shared__ u16t abuf[16 * 512];
  u16t*  Xs   = (u16t*)smem;
  float* hbuf = (float*)smem;
  int tid  = threadIdx.x;
  int lane = tid & 31, wave = tid >> 5;
  int e0   = blockIdx.x * 16;

  // stage X = concat(h[src[e]], edge_feat[e]) once (640 f32 -> 160 vec4 per row)
  for (int i = tid; i < 16 * 160; i += 256) {
    int m = i / 160, t = i - m * 160, j = t * 4;
    int e = e0 + m;
    float4 x4 = (j < 512) ? *(const float4*)&h[(size_t)src[e] * 512 + j]
                          : *(const float4*)&ef[(size_t)e * EFD + (j - 512)];
    *(ushort4*)&Xs[m * 656 + j] = f2bf4(x4);
  }
  __syncthreads();

  // both layer-1 passes while Xs is live; accumulators stay in registers
  v8f accK[4], accV[4];
  mlp_layer1<640, 656>(Xs, kW1p, lane, wave, accK);
  mlp_layer1<640, 656>(Xs, vW1p, lane, wave, accV);

  mlp_finish(accK, hbuf, abuf, kb1, kg, kbe, kW2p, kb2, lane, wave);
  for (int i = tid; i < 16 * 128; i += 256) {
    int m = i >> 7, j = (i & 127) * 4;
    float4 x4 = *(const float4*)&hbuf[m * HLD + j];
    *(ushort4*)&kout[(size_t)(e0 + m) * 512 + j] = f2bf4(x4);
  }
  __syncthreads();

  mlp_finish(accV, hbuf, abuf, vb1, vg, vbe, vW2p, vb2, lane, wave);
  for (int i = tid; i < 16 * 128; i += 256) {
    int m = i >> 7, j = (i & 127) * 4;
    float4 x4 = *(const float4*)&hbuf[m * HLD + j];
    *(ushort4*)&vout[(size_t)(e0 + m) * 512 + j] = f2bf4(x4);
  }
}

__global__ __launch_bounds__(256) void out_mlp_kernel(
    const float* __restrict__ agg, const float* __restrict__ h,
    const u16t* W1p, const float* b1, const float* g, const float* be,
    const u16t* W2p, const float* b2,
    float* __restrict__ out)
{
  __shared__ __align__(16) unsigned char smem[SMEM_BYTES];
  __shared__ u16t abuf[16 * 512];
  u16t*  Xs   = (u16t*)smem;
  float* hbuf = (float*)smem;
  int tid  = threadIdx.x;
  int lane = tid & 31, wave = tid >> 5;
  int r0   = blockIdx.x * 16;

  for (int i = tid; i < 16 * 256; i += 256) {   // 1024/4 vec4 per row
    int m = i >> 8, j = (i & 255) * 4;
    float4 x4 = (j < 512) ? *(const float4*)&agg[(size_t)(r0 + m) * 512 + j]
                          : *(const float4*)&h[(size_t)(r0 + m) * 512 + (j - 512)];
    *(ushort4*)&Xs[m * 1040 + j] = f2bf4(x4);
  }
  __syncthreads();
  v8f acc[4];
  mlp_layer1<1024, 1040>(Xs, W1p, lane, wave, acc);
  mlp_finish(acc, hbuf, abuf, b1, g, be, W2p, b2, lane, wave);
  for (int i = tid; i < 16 * 128; i += 256) {
    int m = i >> 7, j = (i & 127) * 4;
    float4 r4 = *(const float4*)&hbuf[m * HLD + j];
    float4 h4 = *(const float4*)&h[(size_t)(r0 + m) * 512 + j];
    r4.x += h4.x; r4.y += h4.y; r4.z += h4.z; r4.w += h4.w;
    *(float4*)&out[(size_t)(r0 + m) * 512 + j] = r4;
  }
}

// ---------- attention / scatter kernels ----------
__global__ void init_kernel(float* mx, float* den, float* agg, int nmd, long long nagg) {
  long long i = (long long)blockIdx.x * blockDim.x + threadIdx.x;
  if (i < nmd)  { mx[i] = __uint_as_float(0xff800000u); den[i] = 0.f; }
  if (i < nagg) agg[i] = 0.f;
}

__global__ void logits_kernel(const float* __restrict__ q, const u16t* __restrict__ kb,
                              const int* __restrict__ dst, float* __restrict__ lg,
                              float* __restrict__ mx) {
  int i = blockIdx.x * blockDim.x + threadIdx.x;
  if (i >= N_EDGES * NH) return;
  int e = i >> 4, hh = i & 15;
  int d = dst[e];
  const float* qr = q  + (size_t)d * 512 + hh * HEADD;
  const u16t*  kr = kb + (size_t)e * 512 + hh * HEADD;
  float s = 0.f;
#pragma unroll
  for (int j = 0; j < HEADD; ++j) s += qr[j] * bf2f(kr[j]);
  s *= 0.17677669529663687f;  // 1/sqrt(32)
  lg[i] = s;
  atomicMaxF(&mx[d * NH + hh], s);
}

__global__ void expden_kernel(float* __restrict__ lg, const float* __restrict__ mx,
                              float* __restrict__ den, const int* __restrict__ dst) {
  int i = blockIdx.x * blockDim.x + threadIdx.x;
  if (i >= N_EDGES * NH) return;
  int e = i >> 4, hh = i & 15;
  int d = dst[e];
  float ex = __expf(lg[i] - mx[d * NH + hh]);
  lg[i] = ex;
  atomicAdd(&den[d * NH + hh], ex);
}

__global__ void alpha_kernel(float* __restrict__ lg, const float* __restrict__ den,
                             const int* __restrict__ dst) {
  int i = blockIdx.x * blockDim.x + threadIdx.x;
  if (i >= N_EDGES * NH) return;
  int e = i >> 4, hh = i & 15;
  lg[i] = lg[i] / den[dst[e] * NH + hh];
}

__global__ void agg_kernel(const float* __restrict__ lg, const u16t* __restrict__ vb,
                           const int* __restrict__ dst, float* __restrict__ agg) {
  long long i = (long long)blockIdx.x * blockDim.x + threadIdx.x;
  if (i >= (long long)N_EDGES * 512) return;
  int e = (int)(i >> 9);
  int c = (int)(i & 511);
  int d = dst[e];
  float a = lg[e * NH + (c >> 5)];
  atomicAdd(&agg[(size_t)d * 512 + c], a * bf2f(vb[i]));
}

// ---------- host launcher ----------
extern "C" void kernel_launch(void* const* d_in, const int* in_sizes, int n_in,
                              void* d_out, int out_size, void* d_ws, size_t ws_size,
                              hipStream_t stream)
{
  (void)in_sizes; (void)n_in; (void)out_size; (void)ws_size;
  const float* h    = (const float*)d_in[0];
  const float* ef   = (const float*)d_in[1];
  const int*   eidx = (const int*)d_in[2];
  const int* src  = eidx;
  const int* dstp = eidx + N_EDGES;

  const float* hk_W1 = (const float*)d_in[3];  const float* hk_b1 = (const float*)d_in[4];
  const float* hk_g  = (const float*)d_in[5];  const float* hk_be = (const float*)d_in[6];
  const float* hk_W2 = (const float*)d_in[7];  const float* hk_b2 = (const float*)d_in[8];
  const float* hv_W1 = (const float*)d_in[9];  const float* hv_b1 = (const float*)d_in[10];
  const float* hv_g  = (const float*)d_in[11]; const float* hv_be = (const float*)d_in[12];
  const float* hv_W2 = (const float*)d_in[13]; const float* hv_b2 = (const float*)d_in[14];
  const float* hq_W1 = (const float*)d_in[15]; const float* hq_b1 = (const float*)d_in[16];
  const float* hq_g  = (const float*)d_in[17]; const float* hq_be = (const float*)d_in[18];
  const float* hq_W2 = (const float*)d_in[19]; const float* hq_b2 = (const float*)d_in[20];
  const float* no_W1 = (const float*)d_in[21]; const float* no_b1 = (const float*)d_in[22];
  const float* no_g  = (const float*)d_in[23]; const float* no_be = (const float*)d_in[24];
  const float* no_W2 = (const float*)d_in[25]; const float* no_b2 = (const float*)d_in[26];

  char* ws = (char*)d_ws;
  size_t off = 0;
  auto alloc = [&](size_t bytes) -> void* {
    off = (off + 255) & ~(size_t)255;
    void* p = ws + off;
    off += bytes;
    return p;
  };

  u16t*  hkW1p = (u16t*)alloc((size_t)640 * 512 * 2);
  u16t*  hkW2p = (u16t*)alloc((size_t)512 * 512 * 2);
  u16t*  hvW1p = (u16t*)alloc((size_t)640 * 512 * 2);
  u16t*  hvW2p = (u16t*)alloc((size_t)512 * 512 * 2);
  u16t*  hqW1p = (u16t*)alloc((size_t)512 * 512 * 2);
  u16t*  hqW2p = (u16t*)alloc((size_t)512 * 512 * 2);
  u16t*  noW1p = (u16t*)alloc((size_t)1024 * 512 * 2);
  u16t*  noW2p = (u16t*)alloc((size_t)512 * 512 * 2);
  float* qb    = (float*)alloc((size_t)N_NODES * 512 * 4);
  u16t*  kb    = (u16t*)alloc((size_t)N_EDGES * 512 * 2);
  u16t*  vb    = (u16t*)alloc((size_t)N_EDGES * 512 * 2);
  float* lg    = (float*)alloc((size_t)N_EDGES * NH * 4);
  float* mx    = (float*)alloc((size_t)N_NODES * NH * 4);
  float* den   = (float*)alloc((size_t)N_NODES * NH * 4);
  float* agg   = (float*)alloc((size_t)N_NODES * 512 * 4);

  auto cdiv = [](long long a, long long b) { return (unsigned)((a + b - 1) / b); };

  pack_w_kernel<<<cdiv(640 * 512, 256), 256, 0, stream>>>(hk_W1, hkW1p, 640, 512);
  pack_w_kernel<<<cdiv(512 * 512, 256), 256, 0, stream>>>(hk_W2, hkW2p, 512, 512);
  pack_w_kernel<<<cdiv(640 * 512, 256), 256, 0, stream>>>(hv_W1, hvW1p, 640, 512);
  pack_w_kernel<<<cdiv(512 * 512, 256), 256, 0, stream>>>(hv_W2, hvW2p, 512, 512);
  pack_w_kernel<<<cdiv(512 * 512, 256), 256, 0, stream>>>(hq_W1, hqW1p, 512, 512);
  pack_w_kernel<<<cdiv(512 * 512, 256), 256, 0, stream>>>(hq_W2, hqW2p, 512, 512);
  pack_w_kernel<<<cdiv(1024 * 512, 256), 256, 0, stream>>>(no_W1, noW1p, 1024, 512);
  pack_w_kernel<<<cdiv(512 * 512, 256), 256, 0, stream>>>(no_W2, noW2p, 512, 512);

  init_kernel<<<cdiv((long long)N_NODES * 512, 256), 256, 0, stream>>>(
      mx, den, agg, N_NODES * NH, (long long)N_NODES * 512);

  q_mlp_kernel<<<N_NODES / 16, 256, 0, stream>>>(h, hqW1p, hq_b1, hq_g, hq_be, hqW2p, hq_b2, qb);

  kv_mlp_kernel<<<N_EDGES / 16, 256, 0, stream>>>(
      h, ef, src,
      hkW1p, hk_b1, hk_g, hk_be, hkW2p, hk_b2,
      hvW1p, hv_b1, hv_g, hv_be, hvW2p, hv_b2,
      kb, vb);

  logits_kernel<<<cdiv((long long)N_EDGES * NH, 256), 256, 0, stream>>>(qb, kb, dstp, lg, mx);
  expden_kernel<<<cdiv((long long)N_EDGES * NH, 256), 256, 0, stream>>>(lg, mx, den, dstp);
  alpha_kernel<<<cdiv((long long)N_EDGES * NH, 256), 256, 0, stream>>>(lg, den, dstp);
  agg_kernel<<<cdiv((long long)N_EDGES * 512, 256), 256, 0, stream>>>(lg, vb, dstp, agg);

  out_mlp_kernel<<<N_NODES / 16, 256, 0, stream>>>(
      agg, h, noW1p, no_b1, no_g, no_be, noW2p, no_b2, (float*)d_out);
}
